// MoELayer_68822555951675
// MI455X (gfx1250) — matrix operand order repaired and checked by reading, not compiled, not run on hardware
//
#include <hip/hip_runtime.h>
#include <hip/hip_bf16.h>
#include <math.h>
#include <stdint.h>

typedef __attribute__((ext_vector_type(16))) __bf16 v16bf;
typedef __attribute__((ext_vector_type(8)))  float  v8f;

#define B_TOK 4096
#define DDIM  1024
#define HDIM  2048
#define NEXP  8

#define TILE_M 128
#define TILE_N 128
#define TILE_K 32

// ---- CDNA5 async global->LDS copy helpers ---------------------------------
// Copies 32 bytes per lane (two b128 chunks); INST_OFFSET applies to both the
// LDS and global address, so one LDS-addr VGPR + one 64-bit vaddr suffices.
__device__ __forceinline__ void async_copy32(uint32_t lds_off, uint64_t gaddr) {
    asm volatile(
        "global_load_async_to_lds_b128 %0, %1, off\n\t"
        "global_load_async_to_lds_b128 %0, %1, off offset:16"
        :: "v"(lds_off), "v"(gaddr) : "memory");
}

template <int N>
__device__ __forceinline__ void wait_async() {
    asm volatile("s_wait_asynccnt %0" :: "i"(N) : "memory");
}

// ---------------------------------------------------------------- gating ----
__global__ __launch_bounds__(128) void moe_gate_kernel(
    const float* __restrict__ x, const float* __restrict__ Wg,
    const float* __restrict__ bg,
    float* __restrict__ out_logits, float* __restrict__ out_topidx,
    int* __restrict__ counts, int* __restrict__ rank_of,
    int* __restrict__ te_ws, float* __restrict__ gate_rowk)
{
    const int wid  = threadIdx.x >> 5;
    const int lane = threadIdx.x & 31;
    const int row  = blockIdx.x * 4 + wid;
    if (row >= B_TOK) return;

    float acc[NEXP];
#pragma unroll
    for (int e = 0; e < NEXP; ++e) acc[e] = 0.f;

    const float* xr = x + (size_t)row * DDIM;
    for (int d = lane; d < DDIM; d += 32) {
        const float xv = xr[d];
        const float4* wg = (const float4*)(Wg + (size_t)d * NEXP);
        float4 w0 = wg[0], w1 = wg[1];
        acc[0] += xv * w0.x; acc[1] += xv * w0.y;
        acc[2] += xv * w0.z; acc[3] += xv * w0.w;
        acc[4] += xv * w1.x; acc[5] += xv * w1.y;
        acc[6] += xv * w1.z; acc[7] += xv * w1.w;
    }
#pragma unroll
    for (int off = 16; off > 0; off >>= 1) {
#pragma unroll
        for (int e = 0; e < NEXP; ++e) acc[e] += __shfl_xor(acc[e], off, 32);
    }

    if (lane == 0) {
        float lg[NEXP];
#pragma unroll
        for (int e = 0; e < NEXP; ++e) lg[e] = acc[e] + bg[e];
        int e0 = 0;
#pragma unroll
        for (int e = 1; e < NEXP; ++e) if (lg[e] > lg[e0]) e0 = e;
        int e1 = (e0 == 0) ? 1 : 0;
#pragma unroll
        for (int e = 0; e < NEXP; ++e) if (e != e0 && lg[e] > lg[e1]) e1 = e;
        const float v0 = lg[e0], v1 = lg[e1];
        const float p1 = expf(v1 - v0);         // p0 == 1 (v0 is the max)
        const float inv = 1.f / (1.f + p1);

#pragma unroll
        for (int e = 0; e < NEXP; ++e) out_logits[(size_t)row * NEXP + e] = lg[e];
        out_topidx[row * 2 + 0] = (float)e0;
        out_topidx[row * 2 + 1] = (float)e1;

        rank_of[row * 2 + 0] = atomicAdd(&counts[e0], 1);
        rank_of[row * 2 + 1] = atomicAdd(&counts[e1], 1);
        te_ws[row * 2 + 0] = e0;
        te_ws[row * 2 + 1] = e1;
        gate_rowk[row * 2 + 0] = inv;
        gate_rowk[row * 2 + 1] = p1 * inv;
    }
}

__global__ void moe_offsets_kernel(const int* __restrict__ counts,
                                   int* __restrict__ offsets)
{
    if (threadIdx.x == 0 && blockIdx.x == 0) {
        int s = 0;
        for (int e = 0; e < NEXP; ++e) { offsets[e] = s; s += counts[e]; }
        offsets[NEXP] = s;
    }
}

__global__ __launch_bounds__(256) void moe_scatter_kernel(
    const int* __restrict__ te_ws, const int* __restrict__ rank_of,
    const int* __restrict__ offsets,
    int* __restrict__ token_idx, int* __restrict__ slot_of)
{
    const int row = blockIdx.x * blockDim.x + threadIdx.x;
    if (row >= B_TOK) return;
#pragma unroll
    for (int k = 0; k < 2; ++k) {
        const int e = te_ws[row * 2 + k];
        const int slot = offsets[e] + rank_of[row * 2 + k];
        token_idx[slot] = row;
        slot_of[row * 2 + k] = slot;
    }
}

// ----------------------------------------------------- precision converts ---
// x f32 -> bf16, same layout.
__global__ __launch_bounds__(256) void convert_x_kernel(
    const float* __restrict__ x, unsigned short* __restrict__ Xb)
{
    const size_t i = (size_t)(blockIdx.x * 256 + threadIdx.x) * 8;
    float4 a = ((const float4*)(x + i))[0];
    float4 b = ((const float4*)(x + i))[1];
    union { __bf16 h[8]; uint4 u; } o;
    o.h[0] = (__bf16)a.x; o.h[1] = (__bf16)a.y;
    o.h[2] = (__bf16)a.z; o.h[3] = (__bf16)a.w;
    o.h[4] = (__bf16)b.x; o.h[5] = (__bf16)b.y;
    o.h[6] = (__bf16)b.z; o.h[7] = (__bf16)b.w;
    *(uint4*)(Xb + i) = o.u;
}

// W [E][K][N] f32 -> Wt [E][N][K] bf16 (transpose + convert), 32x32 LDS tiles.
__global__ __launch_bounds__(256) void convert_w_kernel(
    const float* __restrict__ W, unsigned short* __restrict__ Wt,
    int K, int N)
{
    __shared__ float tile[32][33];
    const int e  = blockIdx.z;
    const int kt = blockIdx.y * 32;
    const int nt = blockIdx.x * 32;
    const int tid = threadIdx.x;
    {
        const int r = tid >> 3, c = (tid & 7) * 4;
        const float4 v = *(const float4*)(W + ((size_t)e * K + kt + r) * N + nt + c);
        tile[r][c + 0] = v.x; tile[r][c + 1] = v.y;
        tile[r][c + 2] = v.z; tile[r][c + 3] = v.w;
    }
    __syncthreads();
    {
        const int n = tid >> 3, k = (tid & 7) * 4;
        union { __bf16 h[4]; uint2 u; } o;
        o.h[0] = (__bf16)tile[k + 0][n];
        o.h[1] = (__bf16)tile[k + 1][n];
        o.h[2] = (__bf16)tile[k + 2][n];
        o.h[3] = (__bf16)tile[k + 3][n];
        *(uint2*)(Wt + ((size_t)e * N + nt + n) * K + kt + k) = o.u;
    }
}

// ------------------------------------------------------------ FFN GEMM ------
// Double-buffered LDS tiles filled by async DMA; bf16 WMMA, f32 accumulate.
// Buffer parity is a compile-time constant (two stages per loop iteration,
// last two stages peeled) so LDS staging addresses are immediates.

// Issue one 8KB A tile + 8KB B tile into buffer `bi` for K-offset `kbn`.
#define FFN_STAGE_ISSUE(bi, kbn)                                               \
    async_copy32(aoff[bi], agbase + (size_t)(kbn) * 2);                        \
    async_copy32(boff[bi], bgbase + (size_t)(kbn) * 2);

// 8 WMMAs (2 M-frags x 4 N-frags) out of buffer `bi`.
#define FFN_STAGE_COMPUTE(bi)                                                  \
    {                                                                          \
        const __bf16* Abuf = ldsA[bi];                                         \
        const __bf16* Bbuf = ldsB[bi];                                         \
        v16bf afr[2], bfr[4];                                                  \
        _Pragma("unroll")                                                      \
        for (int mt = 0; mt < 2; ++mt) {                                       \
            const __bf16* ap = &Abuf[(wm * 32 + mt * 16 + l16) * TILE_K];      \
            _Pragma("unroll")                                                  \
            for (int i = 0; i < 8; ++i) {                                      \
                afr[mt][i]     = ap[rsel * 16 + i];                            \
                afr[mt][8 + i] = ap[rsel * 16 + 8 + i];                        \
            }                                                                  \
        }                                                                      \
        _Pragma("unroll")                                                      \
        for (int nt = 0; nt < 4; ++nt) {                                       \
            const __bf16* bp =                                                 \
                &Bbuf[(wn * 64 + nt * 16 + l16) * TILE_K + rsel * 16];         \
            _Pragma("unroll")                                                  \
            for (int i = 0; i < 16; ++i) bfr[nt][i] = bp[i];                   \
        }                                                                      \
        _Pragma("unroll")                                                      \
        for (int mt = 0; mt < 2; ++mt)                                         \
            _Pragma("unroll")                                                  \
            for (int nt = 0; nt < 4; ++nt)                                     \
                acc[mt][nt] = __builtin_amdgcn_wmma_f32_16x16x32_bf16(         \
                    false, afr[mt], false, bfr[nt], (short)0, acc[mt][nt],     \
                    false, false);                                             \
    }

#define FFN_BODY(KD)                                                           \
    v8f acc[2][4] = {};                                                        \
    FFN_STAGE_ISSUE(0, 0)                                                      \
    for (int kb = 0; kb < (KD) - 2 * TILE_K; kb += 2 * TILE_K) {               \
        FFN_STAGE_ISSUE(1, kb + TILE_K)                                        \
        wait_async<4>();                                                       \
        __syncthreads();                                                       \
        FFN_STAGE_COMPUTE(0)                                                   \
        __syncthreads();                                                       \
        FFN_STAGE_ISSUE(0, kb + 2 * TILE_K)                                    \
        wait_async<4>();                                                       \
        __syncthreads();                                                       \
        FFN_STAGE_COMPUTE(1)                                                   \
        __syncthreads();                                                       \
    }                                                                          \
    FFN_STAGE_ISSUE(1, (KD) - TILE_K)                                          \
    wait_async<4>();                                                           \
    __syncthreads();                                                           \
    FFN_STAGE_COMPUTE(0)                                                       \
    __syncthreads();                                                           \
    wait_async<0>();                                                           \
    __syncthreads();                                                           \
    FFN_STAGE_COMPUTE(1)

// FFN1: H[slot,0:2048] = relu( Xb[token_idx[slot],:] @ W1t[e]^T + b1[e] )
__global__ __launch_bounds__(256) void moe_ffn1_kernel(
    const unsigned short* __restrict__ Xb16,
    const unsigned short* __restrict__ W1t16,
    const float* __restrict__ b1,
    const int* __restrict__ counts, const int* __restrict__ offsets,
    const int* __restrict__ token_idx, unsigned short* __restrict__ Hb)
{
    __bf16* __restrict__ H = reinterpret_cast<__bf16*>(Hb);

    const int e = blockIdx.z;
    const int cnt = counts[e];
    const int mbase = blockIdx.y * TILE_M;
    if (mbase >= cnt) return;
    const int nbase = blockIdx.x * TILE_N;
    const int slotbase = offsets[e] + mbase;
    int rows_valid = cnt - mbase;
    if (rows_valid > TILE_M) rows_valid = TILE_M;

    __shared__ __bf16 ldsA[2][TILE_M * TILE_K];
    __shared__ __bf16 ldsB[2][TILE_N * TILE_K];

    const int tid  = threadIdx.x;
    const int wid  = tid >> 5;
    const int lane = tid & 31;
    const int wm   = wid & 3;
    const int wn   = wid >> 2;
    const int rsel = lane >> 4;
    const int l16  = lane & 15;

    // Per-thread staging assignment: 32 bytes = half of one 64-byte tile row.
    const int sr = tid >> 1;                 // row (A) / n (B) within tile
    const int sc = (tid & 1) * 32;           // byte offset within row
    const int srcslot = slotbase + (sr < rows_valid ? sr : rows_valid - 1);
    const int trow = token_idx[srcslot];     // clamped gather: EXEC stays full
    const uint64_t agbase =
        (uint64_t)(uintptr_t)(Xb16 + (size_t)trow * DDIM) + sc;
    const uint64_t bgbase =
        (uint64_t)(uintptr_t)(W1t16 + ((size_t)e * HDIM + nbase + sr) * DDIM) + sc;
    uint32_t aoff[2], boff[2];
    aoff[0] = (uint32_t)(uintptr_t)&ldsA[0][0] + tid * 32;
    aoff[1] = (uint32_t)(uintptr_t)&ldsA[1][0] + tid * 32;
    boff[0] = (uint32_t)(uintptr_t)&ldsB[0][0] + tid * 32;
    boff[1] = (uint32_t)(uintptr_t)&ldsB[1][0] + tid * 32;

    FFN_BODY(DDIM)

    // epilogue: bias + relu, bf16 store into compact H
#pragma unroll
    for (int mt = 0; mt < 2; ++mt) {
#pragma unroll
        for (int nt = 0; nt < 4; ++nt) {
            const int col = nbase + wn * 64 + nt * 16 + l16;
            const float bias = b1[e * HDIM + col];
#pragma unroll
            for (int r = 0; r < 8; ++r) {
                const int rl = wm * 32 + mt * 16 + r + rsel * 8;
                if (rl < rows_valid) {
                    float v = acc[mt][nt][r] + bias;
                    v = v > 0.f ? v : 0.f;
                    H[(size_t)(slotbase + rl) * HDIM + col] = (__bf16)v;
                }
            }
        }
    }
}

// FFN2: Y[slot,0:1024] = H[slot,:] @ W2t[e]^T + b2[e]  (f32 out)
__global__ __launch_bounds__(256) void moe_ffn2_kernel(
    const unsigned short* __restrict__ Hb,
    const unsigned short* __restrict__ W2t16,
    const float* __restrict__ b2,
    const int* __restrict__ counts, const int* __restrict__ offsets,
    float* __restrict__ Y)
{
    const int e = blockIdx.z;
    const int cnt = counts[e];
    const int mbase = blockIdx.y * TILE_M;
    if (mbase >= cnt) return;
    const int nbase = blockIdx.x * TILE_N;
    const int slotbase = offsets[e] + mbase;
    int rows_valid = cnt - mbase;
    if (rows_valid > TILE_M) rows_valid = TILE_M;

    __shared__ __bf16 ldsA[2][TILE_M * TILE_K];
    __shared__ __bf16 ldsB[2][TILE_N * TILE_K];

    const int tid  = threadIdx.x;
    const int wid  = tid >> 5;
    const int lane = tid & 31;
    const int wm   = wid & 3;
    const int wn   = wid >> 2;
    const int rsel = lane >> 4;
    const int l16  = lane & 15;

    const int sr = tid >> 1;
    const int sc = (tid & 1) * 32;
    const int srcslot = slotbase + (sr < rows_valid ? sr : rows_valid - 1);
    const uint64_t agbase =
        (uint64_t)(uintptr_t)(Hb + (size_t)srcslot * HDIM) + sc;
    const uint64_t bgbase =
        (uint64_t)(uintptr_t)(W2t16 + ((size_t)e * DDIM + nbase + sr) * HDIM) + sc;
    uint32_t aoff[2], boff[2];
    aoff[0] = (uint32_t)(uintptr_t)&ldsA[0][0] + tid * 32;
    aoff[1] = (uint32_t)(uintptr_t)&ldsA[1][0] + tid * 32;
    boff[0] = (uint32_t)(uintptr_t)&ldsB[0][0] + tid * 32;
    boff[1] = (uint32_t)(uintptr_t)&ldsB[1][0] + tid * 32;

    FFN_BODY(HDIM)

#pragma unroll
    for (int mt = 0; mt < 2; ++mt) {
#pragma unroll
        for (int nt = 0; nt < 4; ++nt) {
            const int col = nbase + wn * 64 + nt * 16 + l16;
            const float bias = b2[e * DDIM + col];
#pragma unroll
            for (int r = 0; r < 8; ++r) {
                const int rl = wm * 32 + mt * 16 + r + rsel * 8;
                if (rl < rows_valid)
                    Y[(size_t)(slotbase + rl) * DDIM + col] = acc[mt][nt][r] + bias;
            }
        }
    }
}

// --------------------------------------------------------------- combine ----
__global__ __launch_bounds__(256) void moe_combine_kernel(
    const float* __restrict__ Y, const int* __restrict__ slot_of,
    const float* __restrict__ gate_rowk, float* __restrict__ out)
{
    const int idx = blockIdx.x * blockDim.x + threadIdx.x;
    const int row = idx >> 8;
    const int c4  = idx & 255;
    const int s0 = slot_of[row * 2 + 0];
    const int s1 = slot_of[row * 2 + 1];
    const float g0 = gate_rowk[row * 2 + 0];
    const float g1 = gate_rowk[row * 2 + 1];
    const float4 a = ((const float4*)(Y + (size_t)s0 * DDIM))[c4];
    const float4 b = ((const float4*)(Y + (size_t)s1 * DDIM))[c4];
    float4 o;
    o.x = g0 * a.x + g1 * b.x;
    o.y = g0 * a.y + g1 * b.y;
    o.z = g0 * a.z + g1 * b.z;
    o.w = g0 * a.w + g1 * b.w;
    ((float4*)(out + (size_t)row * DDIM))[c4] = o;
}

// ---------------------------------------------------------------- launch ----
extern "C" void kernel_launch(void* const* d_in, const int* in_sizes, int n_in,
                              void* d_out, int out_size, void* d_ws, size_t ws_size,
                              hipStream_t stream) {
    const float* x  = (const float*)d_in[0];
    const float* Wg = (const float*)d_in[1];
    const float* bg = (const float*)d_in[2];
    const float* W1 = (const float*)d_in[3];
    const float* b1 = (const float*)d_in[4];
    const float* W2 = (const float*)d_in[5];
    const float* b2 = (const float*)d_in[6];

    float* out        = (float*)d_out;
    float* out_moe    = out;                                  // [4096,1024]
    float* out_logits = out + (size_t)B_TOK * DDIM;           // [4096,8]
    float* out_topidx = out_logits + (size_t)B_TOK * NEXP;    // [4096,2]

    char* ws = (char*)d_ws;
    int*   counts    = (int*)(ws + 0);                        // 8
    int*   offsets   = (int*)(ws + 256);                      // 9
    int*   rank_of   = (int*)(ws + 512);                      // 8192
    int*   te_ws     = (int*)(ws + 512 + 32768);              // 8192
    int*   slot_of   = (int*)(ws + 512 + 2 * 32768);          // 8192
    float* gate_rowk = (float*)(ws + 512 + 3 * 32768);        // 8192
    int*   token_idx = (int*)(ws + 512 + 4 * 32768);          // 8192

    const size_t MB = 1024 * 1024;
    unsigned short* Xb  = (unsigned short*)(ws + 1 * MB);     //  8 MB bf16 x
    unsigned short* W1t = (unsigned short*)(ws + 16 * MB);    // 32 MB bf16 W1^T
    unsigned short* W2t = (unsigned short*)(ws + 48 * MB);    // 32 MB bf16 W2^T
    unsigned short* Hb  = (unsigned short*)(ws + 80 * MB);    // 32 MB bf16 H
    float*          Yb  = (float*)(ws + 112 * MB);            // 32 MB f32 Y

    hipMemsetAsync(counts, 0, NEXP * sizeof(int), stream);

    convert_x_kernel<<<(B_TOK * DDIM / 8) / 256, 256, 0, stream>>>(x, Xb);
    convert_w_kernel<<<dim3(HDIM / 32, DDIM / 32, NEXP), 256, 0, stream>>>(
        W1, W1t, DDIM, HDIM);
    convert_w_kernel<<<dim3(DDIM / 32, HDIM / 32, NEXP), 256, 0, stream>>>(
        W2, W2t, HDIM, DDIM);

    moe_gate_kernel<<<B_TOK / 4, 128, 0, stream>>>(
        x, Wg, bg, out_logits, out_topidx, counts, rank_of, te_ws, gate_rowk);

    moe_offsets_kernel<<<1, 32, 0, stream>>>(counts, offsets);

    moe_scatter_kernel<<<B_TOK / 256, 256, 0, stream>>>(
        te_ws, rank_of, offsets, token_idx, slot_of);

    moe_ffn1_kernel<<<dim3(HDIM / TILE_N, B_TOK / TILE_M, NEXP), 256, 0, stream>>>(
        Xb, W1t, b1, counts, offsets, token_idx, Hb);

    moe_ffn2_kernel<<<dim3(DDIM / TILE_N, B_TOK / TILE_M, NEXP), 256, 0, stream>>>(
        Hb, W2t, b2, counts, offsets, Yb);

    moe_combine_kernel<<<(B_TOK * DDIM / 4) / 256, 256, 0, stream>>>(
        Yb, slot_of, gate_rowk, out_moe);
}
